// PointNetSetAbstraction_35296041239306
// MI455X (gfx1250) — matrix operand order repaired and checked by reading, not compiled, not run on hardware
//
#include <hip/hip_runtime.h>
#include <hip/hip_bf16.h>

typedef __attribute__((ext_vector_type(16))) _Float16 v16h;
typedef __attribute__((ext_vector_type(8)))  _Float16 v8h;
typedef __attribute__((ext_vector_type(8)))  float    v8f;

// Problem constants (from reference setup_inputs)
#define BB 8
#define NN 8192
#define CC 64
#define SS 2048
#define KK 32
#define R2 0.04f
#define EPSV 1e-5f

// Workspace layout (bytes), all 256-aligned
#define OFF_IDX  0u            // int32 [B*S*K]          = 2097152 B
#define OFF_W0   2097152u      // f16  [64][96]          = 12288 B
#define OFF_W1   2109440u      // f16  [64][64]          = 8192 B
#define OFF_W2   2117632u      // f16  [128][64]         = 16384 B
#define OFF_AB   2134016u      // f32  [512] a0 b0 a1 b1 a2 b2

// ---------------------------------------------------------------------------
// Prep: fold BN into per-channel alpha/beta, convert+pad weights to f16
// ---------------------------------------------------------------------------
__global__ __launch_bounds__(256) void prep_kernel(
    const float* W0, const float* b0, const float* g0, const float* be0, const float* m0, const float* v0,
    const float* W1, const float* b1, const float* g1, const float* be1, const float* m1, const float* v1,
    const float* W2, const float* b2, const float* g2, const float* be2, const float* m2, const float* v2,
    _Float16* w0p, _Float16* w1p, _Float16* w2p, float* ab)
{
    int tid = threadIdx.x;
    for (int e = tid; e < 64 * 96; e += 256) {
        int o = e / 96, c = e - o * 96;
        w0p[e] = (c < 67) ? (_Float16)W0[o * 67 + c] : (_Float16)0.0f;
    }
    for (int e = tid; e < 64 * 64; e += 256) w1p[e] = (_Float16)W1[e];
    for (int e = tid; e < 128 * 64; e += 256) w2p[e] = (_Float16)W2[e];
    if (tid < 64) {
        float a = g0[tid] * rsqrtf(v0[tid] + EPSV);
        ab[tid]       = a;
        ab[64 + tid]  = (b0[tid] - m0[tid]) * a + be0[tid];
        float a1 = g1[tid] * rsqrtf(v1[tid] + EPSV);
        ab[128 + tid] = a1;
        ab[192 + tid] = (b1[tid] - m1[tid]) * a1 + be1[tid];
    }
    if (tid < 128) {
        float a2 = g2[tid] * rsqrtf(v2[tid] + EPSV);
        ab[256 + tid] = a2;
        ab[384 + tid] = (b2[tid] - m2[tid]) * a2 + be2[tid];
    }
}

// ---------------------------------------------------------------------------
// FPS: one persistent block per batch. mind[] in LDS; argmax w/ min-index tie
// break (matches jnp.argmax). new_xyz written to d_out[0 .. B*S*3).
// ---------------------------------------------------------------------------
__global__ __launch_bounds__(1024) void fps_kernel(const float* __restrict__ xyz,
                                                   float* __restrict__ out_newxyz)
{
    __shared__ float mind[NN];
    __shared__ int   sel[SS];
    __shared__ float wval[32];
    __shared__ int   widx[32];
    __shared__ int   s_last;

    const int b = blockIdx.x;
    const int tid = threadIdx.x;
    const int lane = tid & 31, wv = tid >> 5;
    const float* bx = xyz + (size_t)b * NN * 3;

    for (int i = tid; i < NN; i += 1024) mind[i] = 1e10f;
    if (tid == 0) { sel[0] = 0; s_last = 0; }
    __syncthreads();

    int last = 0;
    for (int j = 1; j < SS; ++j) {
        float px = bx[last * 3 + 0];
        float py = bx[last * 3 + 1];
        float pz = bx[last * 3 + 2];
        float bv = -1.0f; int bi = 0x7fffffff;
        for (int i = tid; i < NN; i += 1024) {
            float dx = bx[i * 3 + 0] - px;
            float dy = bx[i * 3 + 1] - py;
            float dz = bx[i * 3 + 2] - pz;
            float d = dx * dx + dy * dy + dz * dz;
            float mc = mind[i];
            if (d < mc) { mc = d; mind[i] = mc; }
            if (mc > bv || (mc == bv && i < bi)) { bv = mc; bi = i; }
        }
        #pragma unroll
        for (int off = 16; off; off >>= 1) {
            float ov = __shfl_xor(bv, off, 32);
            int   oi = __shfl_xor(bi, off, 32);
            if (ov > bv || (ov == bv && oi < bi)) { bv = ov; bi = oi; }
        }
        if (lane == 0) { wval[wv] = bv; widx[wv] = bi; }
        __syncthreads();
        if (wv == 0) {
            bv = wval[lane]; bi = widx[lane];
            #pragma unroll
            for (int off = 16; off; off >>= 1) {
                float ov = __shfl_xor(bv, off, 32);
                int   oi = __shfl_xor(bi, off, 32);
                if (ov > bv || (ov == bv && oi < bi)) { bv = ov; bi = oi; }
            }
            if (lane == 0) { sel[j] = bi; s_last = bi; }
        }
        __syncthreads();
        last = s_last;
        __syncthreads();
    }
    // emit new_xyz
    for (int j = tid; j < SS; j += 1024) {
        int i = sel[j];
        size_t o = ((size_t)b * SS + j) * 3;
        out_newxyz[o + 0] = bx[i * 3 + 0];
        out_newxyz[o + 1] = bx[i * 3 + 1];
        out_newxyz[o + 2] = bx[i * 3 + 2];
    }
}

// ---------------------------------------------------------------------------
// Ball query: one wave per query. Candidates (d2 < r2) compacted into LDS as
// u64 keys (d2_bits<<32 | idx) -> ascending-distance, stable-index ordering
// identical to argsort+mask. 32 rounds of wave argmin extraction.
// ---------------------------------------------------------------------------
__global__ __launch_bounds__(256) void ballq_kernel(const float* __restrict__ xyz,
                                                    const float* __restrict__ newxyz,
                                                    int* __restrict__ idxo)
{
    __shared__ unsigned long long cand[8][512];
    __shared__ int ccount[8];

    const int tid = threadIdx.x;
    const int lane = tid & 31, wv = tid >> 5;
    const int q = blockIdx.x * 8 + wv;        // 0 .. B*S-1
    const int b = q >> 11;                    // q / S
    const float* bx = xyz + (size_t)b * NN * 3;

    if (lane == 0) ccount[wv] = 0;
    __syncthreads();

    const float nx = newxyz[q * 3 + 0];
    const float ny = newxyz[q * 3 + 1];
    const float nz = newxyz[q * 3 + 2];

    for (int i = lane; i < NN; i += 32) {
        float dx = bx[i * 3 + 0] - nx;
        float dy = bx[i * 3 + 1] - ny;
        float dz = bx[i * 3 + 2] - nz;
        float d2 = dx * dx + dy * dy + dz * dz;
        if (d2 < R2) {
            int pos = atomicAdd(&ccount[wv], 1);
            if (pos < 512)
                cand[wv][pos] = (((unsigned long long)__float_as_uint(d2)) << 32) |
                                (unsigned long long)(unsigned)i;
        }
    }
    __syncthreads();

    int M = ccount[wv]; if (M > 512) M = 512;
    for (int r = 0; r < KK; ++r) {
        unsigned long long best = ~0ull; int bpos = -1;
        for (int t = lane; t < M; t += 32) {
            unsigned long long v = cand[wv][t];
            if (v < best) { best = v; bpos = t; }
        }
        #pragma unroll
        for (int off = 16; off; off >>= 1) {
            unsigned long long ob = __shfl_xor(best, off, 32);
            int op = __shfl_xor(bpos, off, 32);
            if (ob < best) { best = ob; bpos = op; }
        }
        if (lane == 0) {
            idxo[(size_t)q * KK + r] = (best != ~0ull) ? (int)(unsigned)(best & 0xffffffffu) : -1;
            if (bpos >= 0) cand[wv][bpos] = ~0ull;
        }
    }
}

// ---------------------------------------------------------------------------
// WMMA fragment loaders per CDNA5 16-bit layouts (CDNA5 ISA)
// A 16x32 f16: lanes 0-15 K={0..7,16..23}, lanes 16-31 K={8..15,24..31}
// B 32x16 f16 (weights stored [n][k] row-major): same K split as A
// ---------------------------------------------------------------------------
__device__ inline v16h load_a_frag(const _Float16* tile, int rs, int mt, int kbase, int lane)
{
    int row = (lane & 15) + (mt << 4);
    int kb  = kbase + ((lane >> 4) << 3);
    const _Float16* p = tile + row * rs + kb;
    v8h lo = *(const v8h*)p;          // K kb .. kb+7
    v8h hi = *(const v8h*)(p + 16);   // K kb+16 .. kb+23
    return __builtin_shufflevector(lo, hi, 0,1,2,3,4,5,6,7,8,9,10,11,12,13,14,15);
}

__device__ inline v16h load_b_frag(const _Float16* Wp, int kpad, int nt, int kk, int lane)
{
    int n = (nt << 4) + (lane & 15);
    int k = kk * 32 + ((lane >> 4) << 3);
    const _Float16* p = Wp + n * kpad + k;
    v8h lo = *(const v8h*)p;
    v8h hi = *(const v8h*)(p + 16);
    return __builtin_shufflevector(lo, hi, 0,1,2,3,4,5,6,7,8,9,10,11,12,13,14,15);
}

// Multi-M-tile GEMM: one B-fragment load amortized over MT_N WMMAs.
template<int MT_N, int KSTEPS>
__device__ inline void mm_multi(const _Float16* A, int rsa, const _Float16* Wp, int kpad,
                                int mt0, int nt, int lane, v8f* acc)
{
    #pragma unroll
    for (int kk = 0; kk < KSTEPS; ++kk) {
        v16h b = load_b_frag(Wp, kpad, nt, kk, lane);
        #pragma unroll
        for (int m = 0; m < MT_N; ++m) {
            v16h a = load_a_frag(A, rsa, mt0 + m, kk * 32, lane);
            acc[m] = __builtin_amdgcn_wmma_f32_16x16x32_f16(false, a, false, b,
                                                            (short)0, acc[m], false, false);
        }
    }
}

// C/D layout: VGPR j -> row j + 8*(lane>>4), col = lane&15 (within 16x16 tile)
__device__ inline void store_tile_f16(v8f acc, _Float16* Y, int rsy, int mt, int nt,
                                      const float* alpha, const float* beta, int lane)
{
    int col = (nt << 4) + (lane & 15);
    float al = alpha[col], bt = beta[col];
    int rbase = (mt << 4) + ((lane >> 4) << 3);
    #pragma unroll
    for (int j = 0; j < 8; ++j) {
        float v = fmaxf(acc[j] * al + bt, 0.0f);
        Y[(rbase + j) * rsy + col] = (_Float16)v;
    }
}

// ---------------------------------------------------------------------------
// Fused gather + 3-layer MLP (WMMA) + max over k. 4 groups (M=128) per block.
// Layers 0/1: wave w -> nt = w&3, M-tiles (w>>2)*4 .. +3  (B reused x4).
// Layer 2:    wave w -> nt = w,   M-tiles 0..7            (B reused x8),
// then k-max folded in-register (alpha>0 -> max commutes with affine+relu).
// ---------------------------------------------------------------------------
__global__ __launch_bounds__(256) void mlp_kernel(const float* __restrict__ xyz,
                                                  const float* __restrict__ feat,
                                                  const int* __restrict__ idx,
                                                  const float* __restrict__ newxyz,
                                                  const _Float16* __restrict__ w0p,
                                                  const _Float16* __restrict__ w1p,
                                                  const _Float16* __restrict__ w2p,
                                                  const float* __restrict__ ab,
                                                  float* __restrict__ out)
{
    __shared__ __align__(16) _Float16 X [128 * 96];   // 24 KB
    __shared__ __align__(16) _Float16 Y0[128 * 64];   // 16 KB
    __shared__ __align__(16) _Float16 Y1[128 * 64];   // 16 KB
    __shared__ int   pts[128];
    __shared__ float nx[12];

    const int q0 = blockIdx.x * 4;   // first of 4 groups
    const int b  = q0 >> 11;
    const int tid = threadIdx.x;

    if (tid < 128) { int v = idx[(size_t)q0 * KK + tid]; pts[tid] = (v < 0) ? 0 : v; }
    if (tid < 12)  nx[tid] = newxyz[q0 * 3 + tid];
    __syncthreads();

    // gather + concat + pad -> X[128][96] f16 (4 groups of 32 rows)
    for (int e = tid; e < 128 * 96; e += 256) {
        int r = e / 96, c = e - r * 96;
        int g = r >> 5;
        int p = pts[r];
        float v;
        if (c < 3)       v = xyz[((size_t)b * NN + p) * 3 + c] - nx[g * 3 + c];
        else if (c < 67) v = feat[((size_t)b * NN + p) * CC + (c - 3)];
        else             v = 0.0f;
        X[e] = (_Float16)v;
    }
    __syncthreads();

    const int wv = tid >> 5, lane = tid & 31;

    { // layer0: 67(pad 96) -> 64
        int nt = wv & 3, mt0 = (wv >> 2) * 4;
        v8f acc[4] = {};
        mm_multi<4, 3>(X, 96, w0p, 96, mt0, nt, lane, acc);
        #pragma unroll
        for (int m = 0; m < 4; ++m)
            store_tile_f16(acc[m], Y0, 64, mt0 + m, nt, ab + 0, ab + 64, lane);
    }
    __syncthreads();
    { // layer1: 64 -> 64
        int nt = wv & 3, mt0 = (wv >> 2) * 4;
        v8f acc[4] = {};
        mm_multi<4, 2>(Y0, 64, w1p, 64, mt0, nt, lane, acc);
        #pragma unroll
        for (int m = 0; m < 4; ++m)
            store_tile_f16(acc[m], Y1, 64, mt0 + m, nt, ab + 128, ab + 192, lane);
    }
    __syncthreads();
    { // layer2: 64 -> 128, k-max folded in registers, direct global store
        v8f acc[8] = {};
        mm_multi<8, 2>(Y1, 64, w2p, 64, 0, wv, lane, acc);
        int col = (wv << 4) + (lane & 15);
        float al = ab[256 + col], bt = ab[384 + col];
        #pragma unroll
        for (int g = 0; g < 4; ++g) {
            float m = acc[2 * g][0];
            #pragma unroll
            for (int j = 1; j < 8; ++j) m = fmaxf(m, acc[2 * g][j]);
            #pragma unroll
            for (int j = 0; j < 8; ++j) m = fmaxf(m, acc[2 * g + 1][j]);
            m = fmaxf(m, __shfl_xor(m, 16, 32));     // merge row halves
            float y = fmaxf(m * al + bt, 0.0f);       // alpha>0: max/affine commute
            if (lane < 16)
                out[(size_t)BB * SS * 3 + (size_t)(q0 + g) * 128 + col] = y;
        }
    }
}

// ---------------------------------------------------------------------------
extern "C" void kernel_launch(void* const* d_in, const int* in_sizes, int n_in,
                              void* d_out, int out_size, void* d_ws, size_t ws_size,
                              hipStream_t stream)
{
    (void)in_sizes; (void)n_in; (void)out_size; (void)ws_size;
    const float* xyz  = (const float*)d_in[0];
    const float* feat = (const float*)d_in[1];
    const float* W0 = (const float*)d_in[2],  *b0 = (const float*)d_in[3],
               * g0 = (const float*)d_in[4],  *be0 = (const float*)d_in[5],
               * m0 = (const float*)d_in[6],  *v0 = (const float*)d_in[7];
    const float* W1 = (const float*)d_in[8],  *b1 = (const float*)d_in[9],
               * g1 = (const float*)d_in[10], *be1 = (const float*)d_in[11],
               * m1 = (const float*)d_in[12], *v1 = (const float*)d_in[13];
    const float* W2 = (const float*)d_in[14], *b2 = (const float*)d_in[15],
               * g2 = (const float*)d_in[16], *be2 = (const float*)d_in[17],
               * m2 = (const float*)d_in[18], *v2 = (const float*)d_in[19];

    char* ws = (char*)d_ws;
    int*      idx = (int*)(ws + OFF_IDX);
    _Float16* w0p = (_Float16*)(ws + OFF_W0);
    _Float16* w1p = (_Float16*)(ws + OFF_W1);
    _Float16* w2p = (_Float16*)(ws + OFF_W2);
    float*    ab  = (float*)(ws + OFF_AB);

    float* out = (float*)d_out;           // [B*S*3] new_xyz, then [B*S*128] feats

    prep_kernel<<<1, 256, 0, stream>>>(W0, b0, g0, be0, m0, v0,
                                       W1, b1, g1, be1, m1, v1,
                                       W2, b2, g2, be2, m2, v2,
                                       w0p, w1p, w2p, ab);
    fps_kernel<<<BB, 1024, 0, stream>>>(xyz, out);
    ballq_kernel<<<(BB * SS) / 8, 256, 0, stream>>>(xyz, out, idx);
    mlp_kernel<<<(BB * SS) / 4, 256, 0, stream>>>(xyz, feat, idx, out,
                                                  w0p, w1p, w2p, ab, out);
}
